// DeltaNet_31877247271448
// MI455X (gfx1250) — compile-verified, matrix operand order checked by reading, not run on hardware
//
#include <hip/hip_runtime.h>
#include <math.h>

// ---------------------------------------------------------------------------
// Problem constants (match reference)
// ---------------------------------------------------------------------------
constexpr int B_   = 4;
constexpr int L_   = 4096;
constexpr int DM   = 1024;           // model dim
constexpr int H_   = 4;
constexpr int DK_  = 256;            // head dim (= DV)
constexpr int CH_  = 32;             // chunk
constexpr int NCH_ = L_ / CH_;       // 128 chunks
constexpr int NT   = B_ * L_;        // 16384 token rows
constexpr int GIN  = DM + 16 * H_;   // 1088 gate input dim
constexpr int GH   = 1024;           // gate hidden
constexpr int DSL  = 64;             // dv slice per scan workgroup

typedef __attribute__((ext_vector_type(16))) _Float16 v16h;
typedef __attribute__((ext_vector_type(8)))  float    v8f;

// ---------------------------------------------------------------------------
// WMMA helpers (CDNA5 16x16x32 f16 -> f32)
// ---------------------------------------------------------------------------
__device__ __forceinline__ v8f wmma32(v16h a, v16h b, v8f c) {
  return __builtin_amdgcn_wmma_f32_16x16x32_f16(false, a, false, b,
                                                (short)0, c, false, false);
}

__device__ __forceinline__ int frag_koff(int i, int half) {
  // 16-bit A/B layout: VGPR i holds a K-pair; lanes 16-31 take the second
  // 8-wide K group (ISA 7.12.2, 16-bit A-matrix table). Per lane this is two
  // contiguous 16B chunks -> compiler emits paired global_load_b128.
  return (i < 4) ? (half * 8 + i * 2) : (16 + half * 8 + (i - 4) * 2);
}

// A-fragment (16x32, MxK) from row-major source (ld elems); also used for the
// B operand when B is stored N-major (i.e. W[n][k] row-major == B^T).
__device__ __forceinline__ v16h load_frag_km(const _Float16* p, int ld) {
  int lane = threadIdx.x & 31;
  int m = lane & 15, half = lane >> 4;
  v16h f;
#pragma unroll
  for (int i = 0; i < 8; ++i) {
    int kk = frag_koff(i, half);
    f[2 * i]     = p[(size_t)m * ld + kk];
    f[2 * i + 1] = p[(size_t)m * ld + kk + 1];
  }
  return f;
}

// B-fragment (32x16, KxN) from row-major (K,N) source (ld = row stride).
// Also used as a transposed-A loader: element (m,kk) = src[kk*ld + m].
__device__ __forceinline__ v16h load_frag_kn(const _Float16* p, int ld) {
  int lane = threadIdx.x & 31;
  int n = lane & 15, half = lane >> 4;
  v16h f;
#pragma unroll
  for (int i = 0; i < 8; ++i) {
    int kk = frag_koff(i, half);
    f[2 * i]     = p[(size_t)kk * ld + n];
    f[2 * i + 1] = p[(size_t)(kk + 1) * ld + n];
  }
  return f;
}

// ---------------------------------------------------------------------------
// Elementwise / small kernels
// ---------------------------------------------------------------------------
__global__ void k_cast16(const float* __restrict__ x, _Float16* __restrict__ y,
                         size_t n) {
  for (size_t i = (size_t)blockIdx.x * blockDim.x + threadIdx.x; i < n;
       i += (size_t)gridDim.x * blockDim.x)
    y[i] = (_Float16)x[i];
}

// Depthwise causal conv over L with optional SiLU. x,y: (B,L,DM); w: (DM,Kf)
__global__ void k_dwconv(const float* __restrict__ x, const float* __restrict__ w,
                         float* __restrict__ y, int Kf, int act) {
  size_t total = (size_t)NT * DM;
  for (size_t idx = (size_t)blockIdx.x * blockDim.x + threadIdx.x; idx < total;
       idx += (size_t)gridDim.x * blockDim.x) {
    int c = (int)(idx % DM);
    size_t bl = idx / DM;
    int l = (int)(bl % L_);
    size_t brow = bl - l;   // b*L
    float s = 0.f;
    for (int t = 0; t < Kf; ++t) {
      int ls = l - (Kf - 1) + t;
      if (ls >= 0) s += x[(brow + ls) * DM + c] * w[c * Kf + t];
    }
    if (act) s = s / (1.f + expf(-s));   // SiLU
    y[idx] = s;
  }
}

// beta = sigmoid(hidden @ Wb^T); Wb: (H, DM)
__global__ void k_beta(const float* __restrict__ h, const float* __restrict__ Wb,
                       float* __restrict__ beta) {
  int r = blockIdx.x;                     // token row
  int t = threadIdx.x;                    // 128
  int j = t >> 5, lane = t & 31;
  float s = 0.f;
  for (int d = lane; d < DM; d += 32) s += h[(size_t)r * DM + d] * Wb[j * DM + d];
#pragma unroll
  for (int off = 16; off >= 1; off >>= 1) s += __shfl_xor(s, off, 32);
  if (lane == 0) beta[(size_t)r * H_ + j] = 1.f / (1.f + expf(-s));
}

// logits = h1 @ W2^T + b2; W2: (16, GH)
__global__ void k_logits(const float* __restrict__ h1, const float* __restrict__ W2,
                         const float* __restrict__ b2, float* __restrict__ lg) {
  int r = blockIdx.x;
  int t = threadIdx.x;                    // 128
  int j = t >> 3, sub = t & 7;
  float s = 0.f;
  for (int d = sub; d < GH; d += 8) s += h1[(size_t)r * GH + d] * W2[j * GH + d];
#pragma unroll
  for (int off = 4; off >= 1; off >>= 1) s += __shfl_xor(s, off, 32);
  if (sub == 0) lg[(size_t)r * 16 + j] = s + b2[j];
}

// gate_in assembly: [hidden | stats(short) | stats(long) | stats(delta) | stats(v)]
__global__ void k_stats(const float* __restrict__ hidden, const float* __restrict__ ps,
                        const float* __restrict__ pl, const float* __restrict__ pd,
                        const float* __restrict__ vd, float* __restrict__ gin) {
  int r = blockIdx.x;
  int tid = threadIdx.x;                  // 128
  size_t rb = (size_t)r * DM;
  size_t gb = (size_t)r * GIN;
  for (int e = tid; e < DM; e += 128) gin[gb + e] = hidden[rb + e];
  int combo = (tid >> 5) * 4 + ((tid & 31) >> 3);   // 16 combos = (p,h)
  int sub = tid & 7;
  int p = combo >> 2, hh = combo & 3;
  const float* src = (p == 0) ? ps : (p == 1) ? pl : (p == 2) ? pd : vd;
  src += rb + hh * DK_;
  float s1 = 0.f, s2 = 0.f, sa = 0.f;
  for (int i = sub; i < DK_; i += 8) {
    float x = src[i];
    s1 += x; s2 += x * x; sa += fabsf(x);
  }
#pragma unroll
  for (int off = 4; off >= 1; off >>= 1) {
    s1 += __shfl_xor(s1, off, 32);
    s2 += __shfl_xor(s2, off, 32);
    sa += __shfl_xor(sa, off, 32);
  }
  if (sub == 0) {
    float mean = s1 / 256.f;
    float* o = gin + gb + DM + p * 16 + hh * 4;
    o[0] = mean;
    o[1] = s2 / 256.f - mean * mean;
    o[2] = sa / 256.f;
    o[3] = sqrtf(s2);
  }
}

// softmax(tau)/floor gating + path mix + per-head RMSNorm -> o16 for final GEMM
__global__ void k_combine(const float* __restrict__ lg, const float* __restrict__ ltemp,
                          const float* __restrict__ fpar, const float* __restrict__ ps,
                          const float* __restrict__ pl, const float* __restrict__ pd,
                          const float* __restrict__ vd, const float* __restrict__ onw,
                          _Float16* __restrict__ o16) {
  __shared__ float wts[16];
  __shared__ float ssq[4];
  int r = blockIdx.x;
  int tid = threadIdx.x;                  // 256
  if (tid < 16) {
    float tau = logf(1.f + expf(ltemp[tid])) + 0.25f;
    float z = lg[(size_t)r * 16 + tid] / tau;
    float m = z;
#pragma unroll
    for (int off = 2; off >= 1; off >>= 1) m = fmaxf(m, __shfl_xor(m, off, 32));
    float e = expf(z - m), sum = e;
#pragma unroll
    for (int off = 2; off >= 1; off >>= 1) sum += __shfl_xor(sum, off, 32);
    float prob = e / sum;
    float fl = fminf(0.005f + 0.07f / (1.f + expf(-fpar[tid])), 0.25f);
    float sfl = fl;
#pragma unroll
    for (int off = 2; off >= 1; off >>= 1) sfl += __shfl_xor(sfl, off, 32);
    sfl = fminf(sfl, 0.99f);
    float cl = fmaxf(prob, fl / sfl * 0.99f + 1e-9f);
    float cs = cl;
#pragma unroll
    for (int off = 2; off >= 1; off >>= 1) cs += __shfl_xor(cs, off, 32);
    wts[tid] = cl / (cs + 1e-8f);
  }
  if (tid < 4) ssq[tid] = 0.f;
  __syncthreads();
  float ov[4];
#pragma unroll
  for (int i = 0; i < 4; ++i) {
    int e = tid + 256 * i;
    int hh = e >> 8;
    size_t idx = (size_t)r * DM + e;
    float o = wts[hh * 4 + 0] * ps[idx] + wts[hh * 4 + 1] * pl[idx] +
              wts[hh * 4 + 2] * pd[idx] + wts[hh * 4 + 3] * vd[idx];
    ov[i] = o;
    atomicAdd(&ssq[hh], o * o);
  }
  __syncthreads();
#pragma unroll
  for (int i = 0; i < 4; ++i) {
    int e = tid + 256 * i;
    int hh = e >> 8, d = e & 255;
    float rms = rsqrtf(ssq[hh] / 256.f + 1e-5f);
    o16[(size_t)r * DM + e] = (_Float16)(ov[i] * rms * onw[d]);
  }
}

// ---------------------------------------------------------------------------
// Generic NT GEMM: C(M,N) f32 = A(M,K) f16 @ W(N,K)^T f16 [+bias][gelu]
// Block = 256 thr = 8 waves (2x4). Wave tile 32x64 (2 A-frags x 4 B-frags
// -> 8 wmma per k-step), block tile 64x256. M%64==0, N%256==0, K%32==0 for
// every launch below, so no edge handling (EXEC all-ones around WMMA).
// ---------------------------------------------------------------------------
__global__ void k_gemm_nt(const _Float16* __restrict__ A, const _Float16* __restrict__ W,
                          const float* __restrict__ bias, float* __restrict__ C,
                          int M, int N, int K, int act) {
  int wave = threadIdx.x >> 5;
  int lane = threadIdx.x & 31;
  int wm = wave >> 2, wn = wave & 3;
  int m0 = blockIdx.y * 64 + wm * 32;
  int n0 = blockIdx.x * 256 + wn * 64;
  v8f acc[2][4] = {};
  for (int k0 = 0; k0 < K; k0 += 32) {
    if (k0 + 64 < K) {
      // pull the K-tile two steps ahead toward the WGP (global_prefetch_b8)
      __builtin_prefetch(A + (size_t)m0 * K + k0 + 64, 0, 3);
      __builtin_prefetch(W + (size_t)n0 * K + k0 + 64, 0, 3);
    }
    v16h a0 = load_frag_km(A + (size_t)m0 * K + k0, K);
    v16h a1 = load_frag_km(A + (size_t)(m0 + 16) * K + k0, K);
    v16h b0 = load_frag_km(W + (size_t)n0 * K + k0, K);
    v16h b1 = load_frag_km(W + (size_t)(n0 + 16) * K + k0, K);
    v16h b2 = load_frag_km(W + (size_t)(n0 + 32) * K + k0, K);
    v16h b3 = load_frag_km(W + (size_t)(n0 + 48) * K + k0, K);
    acc[0][0] = wmma32(a0, b0, acc[0][0]);
    acc[0][1] = wmma32(a0, b1, acc[0][1]);
    acc[0][2] = wmma32(a0, b2, acc[0][2]);
    acc[0][3] = wmma32(a0, b3, acc[0][3]);
    acc[1][0] = wmma32(a1, b0, acc[1][0]);
    acc[1][1] = wmma32(a1, b1, acc[1][1]);
    acc[1][2] = wmma32(a1, b2, acc[1][2]);
    acc[1][3] = wmma32(a1, b3, acc[1][3]);
  }
  int nn = lane & 15, half = lane >> 4;
#pragma unroll
  for (int ti = 0; ti < 2; ++ti) {
#pragma unroll
    for (int tj = 0; tj < 4; ++tj) {
#pragma unroll
      for (int rr = 0; rr < 8; ++rr) {
        int row = m0 + ti * 16 + rr + 8 * half;
        int col = n0 + tj * 16 + nn;
        float x = acc[ti][tj][rr];
        if (bias) x += bias[col];
        if (act == 1) x = 0.5f * x * (1.f + erff(x * 0.70710678f));
        C[(size_t)row * N + col] = x;
      }
    }
  }
}

// ---------------------------------------------------------------------------
// Delta rule pass 1 (chunk-parallel): per (b,h,chunk) compute
//   qn,kn = l2norm(q),l2norm(k); kb=kn*beta; vb=v*beta
//   Tinv = (I - tril(-(kb kn^T),-1))^{-1}  (unit-lower forward substitution)
//   u = Tinv @ vb, w = Tinv @ kb, attn = tril(qn kn^T)
// Block = 128 thr (4 waves). Dynamic LDS = 75776 B.
// ---------------------------------------------------------------------------
__global__ void k_delta_pass1(const float* __restrict__ qc, const float* __restrict__ kc,
                              const float* __restrict__ vc, const float* __restrict__ beta,
                              _Float16* __restrict__ qn_g, _Float16* __restrict__ kn_g,
                              _Float16* __restrict__ u_g, _Float16* __restrict__ w_g,
                              _Float16* __restrict__ attn_g) {
  extern __shared__ char smem[];
  _Float16* qn  = (_Float16*)smem;          // 32x256
  _Float16* kn  = qn + 32 * DK_;
  _Float16* kb  = kn + 32 * DK_;
  _Float16* vb  = kb + 32 * DK_;
  _Float16* T16 = vb + 32 * DK_;            // 32x32
  float*    As  = (float*)(T16 + 32 * 32);  // 32x32
  float*    T32 = As + 32 * 32;             // 32x32

  int cid = blockIdx.x;                     // bh*NCH + ch
  int ch = cid % NCH_, bh = cid / NCH_;
  int h = bh % H_, b = bh / H_;
  int l0 = ch * CH_;
  int tid = threadIdx.x, wave = tid >> 5, lane = tid & 31;

  // rows: l2norm + beta scaling
  for (int r = wave; r < 32; r += 4) {
    size_t row = ((size_t)(b * L_ + l0 + r)) * DM + h * DK_;
    float bq = beta[(size_t)(b * L_ + l0 + r) * H_ + h];
    float qv[8], kv[8];
    float sq = 0.f, sk = 0.f;
#pragma unroll
    for (int i = 0; i < 8; ++i) {
      float x = qc[row + lane + 32 * i]; qv[i] = x; sq += x * x;
      float y = kc[row + lane + 32 * i]; kv[i] = y; sk += y * y;
    }
#pragma unroll
    for (int off = 16; off >= 1; off >>= 1) {
      sq += __shfl_xor(sq, off, 32);
      sk += __shfl_xor(sk, off, 32);
    }
    float rq = rsqrtf(sq + 1e-6f), rk = rsqrtf(sk + 1e-6f);
#pragma unroll
    for (int i = 0; i < 8; ++i) {
      int cc = lane + 32 * i;
      float knv = kv[i] * rk;
      qn[r * DK_ + cc] = (_Float16)(qv[i] * rq);
      kn[r * DK_ + cc] = (_Float16)knv;
      kb[r * DK_ + cc] = (_Float16)(knv * bq);
      vb[r * DK_ + cc] = (_Float16)(vc[row + cc] * bq);
    }
  }
  __syncthreads();

  int nn = lane & 15, half = lane >> 4;

  // As = strict-lower of -(kb @ kn^T)   (WMMA, 2x2 tiles over K=256)
  {
    int ti = wave >> 1, tj = wave & 1;
    v8f acc = {};
    for (int k0 = 0; k0 < DK_; k0 += 32) {
      v16h a = load_frag_km(kb + (ti * 16) * DK_ + k0, DK_);
      v16h bb = load_frag_km(kn + (tj * 16) * DK_ + k0, DK_);
      acc = wmma32(a, bb, acc);
    }
#pragma unroll
    for (int rr = 0; rr < 8; ++rr) {
      int i = ti * 16 + rr + 8 * half, j = tj * 16 + nn;
      As[i * 32 + j] = (i > j) ? -acc[rr] : 0.f;
    }
  }
  for (int e = tid; e < 1024; e += 128) T32[e] = ((e >> 5) == (e & 31)) ? 1.f : 0.f;
  __syncthreads();

  // forward substitution: T32[i,:] = e_i + sum_{j<i} As[i,j] * T32[j,:]
  for (int i = 1; i < 32; ++i) {
    if (tid < 32) {
      float a2 = 0.f;
      for (int j = 0; j < i; ++j) a2 += As[i * 32 + j] * T32[j * 32 + tid];
      T32[i * 32 + tid] += a2;
    }
    __syncthreads();
  }
  for (int e = tid; e < 1024; e += 128) T16[e] = (_Float16)T32[e];
  __syncthreads();

  size_t ob = (size_t)cid * 32 * DK_;

  // u = T @ vb, w = T @ kb  (32 tiles of 16x16, K=32)
  for (int t = wave; t < 32; t += 4) {
    int ti = t >> 4, tj = t & 15;
    int m0 = ti * 16, n0 = tj * 16;
    v16h af = load_frag_km(T16 + m0 * 32, 32);
    v16h bu = load_frag_kn(vb + n0, DK_);
    v16h bw = load_frag_kn(kb + n0, DK_);
    v8f au = {}, aw = {};
    au = wmma32(af, bu, au);
    aw = wmma32(af, bw, aw);
#pragma unroll
    for (int rr = 0; rr < 8; ++rr) {
      int row = m0 + rr + 8 * half, col = n0 + nn;
      u_g[ob + row * DK_ + col] = (_Float16)au[rr];
      w_g[ob + row * DK_ + col] = (_Float16)aw[rr];
    }
  }

  // attn = tril(qn @ kn^T) incl diagonal
  {
    int ti = wave >> 1, tj = wave & 1;
    v8f acc = {};
    for (int k0 = 0; k0 < DK_; k0 += 32) {
      v16h a = load_frag_km(qn + (ti * 16) * DK_ + k0, DK_);
      v16h bb = load_frag_km(kn + (tj * 16) * DK_ + k0, DK_);
      acc = wmma32(a, bb, acc);
    }
    size_t ab = (size_t)cid * 32 * 32;
#pragma unroll
    for (int rr = 0; rr < 8; ++rr) {
      int i = ti * 16 + rr + 8 * half, j = tj * 16 + nn;
      attn_g[ab + i * 32 + j] = (_Float16)((i >= j) ? acc[rr] : 0.f);
    }
  }

  for (int e = tid; e < 32 * DK_; e += 128) {
    qn_g[ob + e] = qn[e];
    kn_g[ob + e] = kn[e];
  }
}

// ---------------------------------------------------------------------------
// Delta rule pass 2 (sequential scan over chunks), DV split into 4 slices of 64.
// State slice S (256 x 64) kept f32 in LDS with an f16 mirror for WMMA.
// Block = 256 thr (8 waves). Dynamic LDS = 102400 B.
// ---------------------------------------------------------------------------
__global__ void k_delta_scan(const _Float16* __restrict__ qn_g, const _Float16* __restrict__ kn_g,
                             const _Float16* __restrict__ u_g, const _Float16* __restrict__ w_g,
                             const _Float16* __restrict__ attn_g, float* __restrict__ od) {
  extern __shared__ char smem[];
  float*    S32 = (float*)smem;                    // 256 x 64
  _Float16* S16 = (_Float16*)(S32 + 256 * DSL);    // 256 x 64
  _Float16* U16 = S16 + 256 * DSL;                 // 32 x 64

  int blk = blockIdx.x;
  int s = blk & 3, bh = blk >> 2;
  int h = bh % H_, b = bh / H_;
  int tid = threadIdx.x, wave = tid >> 5, lane = tid & 31;
  int nn = lane & 15, half = lane >> 4;
  int wm = wave >> 2, wn = wave & 3;               // phase-A tile (2x4)
  int m0 = wm * 16, n0 = wn * 16;

  for (int e = tid; e < 256 * DSL; e += 256) { S32[e] = 0.f; S16[e] = (_Float16)0.f; }
  __syncthreads();

  for (int ch = 0; ch < NCH_; ++ch) {
    size_t cb = ((size_t)(bh * NCH_ + ch)) * 32 * DK_;

    // phase A: x = q @ S, y = w @ S   (reads S16 of previous chunk)
    v8f xa = {}, ya = {};
    for (int k0 = 0; k0 < DK_; k0 += 32) {
      v16h aq = load_frag_km(qn_g + cb + m0 * DK_ + k0, DK_);
      v16h aw = load_frag_km(w_g + cb + m0 * DK_ + k0, DK_);
      v16h bs = load_frag_kn(S16 + k0 * DSL + n0, DSL);
      xa = wmma32(aq, bs, xa);
      ya = wmma32(aw, bs, ya);
    }
    // u_adj = u - w@S  -> U16
#pragma unroll
    for (int rr = 0; rr < 8; ++rr) {
      int row = m0 + rr + 8 * half, col = n0 + nn;
      float uu = (float)u_g[cb + row * DK_ + s * DSL + col] - ya[rr];
      U16[row * DSL + col] = (_Float16)uu;
    }
    __syncthreads();

    // o = q@S + attn @ u_adj
    {
      v16h aa = load_frag_km(attn_g + (size_t)(bh * NCH_ + ch) * 1024 + m0 * 32, 32);
      v16h bu = load_frag_kn(U16 + n0, DSL);
      xa = wmma32(aa, bu, xa);
    }
#pragma unroll
    for (int rr = 0; rr < 8; ++rr) {
      int row = m0 + rr + 8 * half, col = n0 + nn;
      int l = ch * CH_ + row;
      od[((size_t)(b * L_ + l)) * DM + h * DK_ + s * DSL + col] = xa[rr];
    }

    // phase B: S += k^T @ u_adj   (each wave: 2 dk-tiles x 4 dv-tiles)
#pragma unroll
    for (int a = 0; a < 2; ++a) {
      int mi = (wave * 2 + a) * 16;
      v16h ak = load_frag_kn(kn_g + cb + mi, DK_);   // k^T fragment
#pragma unroll
      for (int tj = 0; tj < 4; ++tj) {
        int nj = tj * 16;
        v16h bu = load_frag_kn(U16 + nj, DSL);
        v8f c;
#pragma unroll
        for (int rr = 0; rr < 8; ++rr)
          c[rr] = S32[(mi + rr + 8 * half) * DSL + nj + nn];
        c = wmma32(ak, bu, c);
#pragma unroll
        for (int rr = 0; rr < 8; ++rr) {
          int e = (mi + rr + 8 * half) * DSL + nj + nn;
          S32[e] = c[rr];
          S16[e] = (_Float16)c[rr];
        }
      }
    }
    __syncthreads();   // next chunk's phase A must see updated S16
  }
}

// ---------------------------------------------------------------------------
// Host-side orchestration
// ---------------------------------------------------------------------------
extern "C" void kernel_launch(void* const* d_in, const int* in_sizes, int n_in,
                              void* d_out, int out_size, void* d_ws, size_t ws_size,
                              hipStream_t stream) {
  (void)in_sizes; (void)n_in; (void)out_size; (void)ws_size;

  const float* hidden = (const float*)d_in[0];
  const float* Wq = (const float*)d_in[1];
  const float* Wk = (const float*)d_in[2];
  const float* Wv = (const float*)d_in[3];
  const float* Wb = (const float*)d_in[4];
  const float* cqw = (const float*)d_in[5];
  const float* ckw = (const float*)d_in[6];
  const float* cvw = (const float*)d_in[7];
  const float* fsw = (const float*)d_in[8];
  const float* flw = (const float*)d_in[9];
  const float* ltemp = (const float*)d_in[10];
  const float* fpar = (const float*)d_in[11];
  const float* gW1 = (const float*)d_in[12];
  const float* gb1 = (const float*)d_in[13];
  const float* gW2 = (const float*)d_in[14];
  const float* gb2 = (const float*)d_in[15];
  const float* onw = (const float*)d_in[16];
  const float* Wo = (const float*)d_in[17];
  float* out = (float*)d_out;

  const size_t NTD = (size_t)NT * DM;    // 16.7M elems

  char* ws = (char*)d_ws;
  size_t off = 0;
  auto alloc = [&](size_t bytes) -> char* {
    char* p = ws + off;
    off = (off + bytes + 255) & ~(size_t)255;
    return p;
  };

  _Float16* h16  = (_Float16*)alloc(NTD * 2);
  _Float16* wq16 = (_Float16*)alloc((size_t)DM * DM * 2);
  _Float16* wk16 = (_Float16*)alloc((size_t)DM * DM * 2);
  _Float16* wv16 = (_Float16*)alloc((size_t)DM * DM * 2);
  _Float16* wo16 = (_Float16*)alloc((size_t)DM * DM * 2);
  _Float16* gw1_16 = (_Float16*)alloc((size_t)GH * GIN * 2);
  float* q32 = (float*)alloc(NTD * 4);   // later reused: fir_short
  float* k32 = (float*)alloc(NTD * 4);   // later reused: fir_long
  float* v32 = (float*)alloc(NTD * 4);   // later reused: path_delta
  float* qc  = (float*)alloc(NTD * 4);   // later reused: h1
  float* kc  = (float*)alloc(NTD * 4);   // later reused: logits, o16
  float* vc  = (float*)alloc(NTD * 4);   // stays live (v_direct)
  float* beta = (float*)alloc((size_t)NT * H_ * 4);
  _Float16* qn = (_Float16*)alloc((size_t)B_ * H_ * L_ * DK_ * 2);
  _Float16* kn = (_Float16*)alloc((size_t)B_ * H_ * L_ * DK_ * 2);
  _Float16* ug = (_Float16*)alloc((size_t)B_ * H_ * L_ * DK_ * 2);
  _Float16* wg = (_Float16*)alloc((size_t)B_ * H_ * L_ * DK_ * 2);
  _Float16* attn = (_Float16*)alloc((size_t)B_ * H_ * NCH_ * 1024 * 2);
  float* gin32 = (float*)alloc((size_t)NT * GIN * 4);
  _Float16* gin16 = (_Float16*)alloc((size_t)NT * GIN * 2);
  // aliases (lifetimes disjoint):
  float* firs = q32;
  float* firl = k32;
  float* pdelta = v32;
  float* h1 = qc;
  float* logits = kc;
  _Float16* o16 = (_Float16*)(kc + 4 * 1024 * 1024);   // 16MB into dead kc

  const int P1_LDS = 4 * 32 * DK_ * 2 + 32 * 32 * 2 + 2 * 32 * 32 * 4;   // 75776
  const int SC_LDS = 256 * DSL * 4 + 256 * DSL * 2 + 32 * DSL * 2;       // 102400
  (void)hipFuncSetAttribute((const void*)k_delta_pass1,
                            hipFuncAttributeMaxDynamicSharedMemorySize, P1_LDS);
  (void)hipFuncSetAttribute((const void*)k_delta_scan,
                            hipFuncAttributeMaxDynamicSharedMemorySize, SC_LDS);

  // 1. f32 -> f16 casts
  k_cast16<<<4096, 256, 0, stream>>>(hidden, h16, NTD);
  k_cast16<<<1024, 256, 0, stream>>>(Wq, wq16, (size_t)DM * DM);
  k_cast16<<<1024, 256, 0, stream>>>(Wk, wk16, (size_t)DM * DM);
  k_cast16<<<1024, 256, 0, stream>>>(Wv, wv16, (size_t)DM * DM);
  k_cast16<<<1024, 256, 0, stream>>>(Wo, wo16, (size_t)DM * DM);
  k_cast16<<<1024, 256, 0, stream>>>(gW1, gw1_16, (size_t)GH * GIN);

  // 2. q/k/v projections (WMMA GEMM, block tile 64x256)
  dim3 gg(DM / 256, NT / 64);
  k_gemm_nt<<<gg, 256, 0, stream>>>(h16, wq16, nullptr, q32, NT, DM, DM, 0);
  k_gemm_nt<<<gg, 256, 0, stream>>>(h16, wk16, nullptr, k32, NT, DM, DM, 0);
  k_gemm_nt<<<gg, 256, 0, stream>>>(h16, wv16, nullptr, v32, NT, DM, DM, 0);

  // 3. beta
  k_beta<<<NT, 128, 0, stream>>>(hidden, Wb, beta);

  // 4. short causal conv + SiLU
  k_dwconv<<<65536, 256, 0, stream>>>(q32, cqw, qc, 4, 1);
  k_dwconv<<<65536, 256, 0, stream>>>(k32, ckw, kc, 4, 1);
  k_dwconv<<<65536, 256, 0, stream>>>(v32, cvw, vc, 4, 1);

  // 5. FIR paths (q32/k32 now dead -> reuse)
  k_dwconv<<<65536, 256, 0, stream>>>(vc, fsw, firs, 5, 0);
  k_dwconv<<<65536, 256, 0, stream>>>(vc, flw, firl, 63, 0);

  // 6. delta rule: chunk-parallel precompute, then DV-sliced sequential scan
  k_delta_pass1<<<B_ * H_ * NCH_, 128, P1_LDS, stream>>>(qc, kc, vc, beta,
                                                         qn, kn, ug, wg, attn);
  k_delta_scan<<<B_ * H_ * 4, 256, SC_LDS, stream>>>(qn, kn, ug, wg, attn, pdelta);

  // 7. gate input (hidden + 4x path stats)
  k_stats<<<NT, 128, 0, stream>>>(hidden, firs, firl, pdelta, vc, gin32);
  k_cast16<<<4096, 256, 0, stream>>>(gin32, gin16, (size_t)NT * GIN);

  // 8. gate MLP: h1 = gelu(gin @ W1^T + b1)  (WMMA, K=1088)
  dim3 g2(GH / 256, NT / 64);
  k_gemm_nt<<<g2, 256, 0, stream>>>(gin16, gw1_16, gb1, h1, NT, GH, GIN, 1);
  k_logits<<<NT, 128, 0, stream>>>(h1, gW2, gb2, logits);

  // 9. gating + mix + RMSNorm -> f16
  k_combine<<<NT, 256, 0, stream>>>(logits, ltemp, fpar, firs, firl, pdelta, vc,
                                    onw, o16);

  // 10. output projection (WMMA)
  k_gemm_nt<<<gg, 256, 0, stream>>>(o16, wo16, nullptr, out, NT, DM, DM, 0);
}
